// Model_66125316489383
// MI455X (gfx1250) — compile-verified, matrix operand order checked by reading, not run on hardware
//
#include <hip/hip_runtime.h>
#include <hip/hip_bf16.h>

// ---------------------------------------------------------------------------
// MI455X plan:
//   s_final[i,j] = s0[i,j] * r[i] * c[j] where s0 = exp(f0 f1^T / T) + eps.
//   Sinkhorn even step:  c = 1 / (s0^T r);  odd step:  r = 1 / (s0 c).
//   s0 (64 MB) is written once into d_out and stays resident in the 192 MB L2
//   for the 10 mat-vec passes. Similarity GEMM uses v_wmma_f32_16x16x32_f16
//   (f16 in / f32 accumulate; inputs are unit-norm so f16 is safe).
// ---------------------------------------------------------------------------

#define NN   4096
#define EMB  300
#define KH   320          // K padded to 10 x 32 for the f16 WMMA k-loop
#define ROWS 16           // rows per projector block

typedef _Float16 v16h __attribute__((ext_vector_type(16)));
typedef float    v8f  __attribute__((ext_vector_type(8)));

// ---------------- Kernel 1: projector + l2norm -> padded f16 rows ----------
__global__ __launch_bounds__(320) void proj_kernel(
    const float* __restrict__ h0, const float* __restrict__ h1,
    const float* __restrict__ W1, const float* __restrict__ b1,
    const float* __restrict__ W2, const float* __restrict__ b2,
    _Float16* __restrict__ f0h, _Float16* __restrict__ f1h)
{
  __shared__ float hs[ROWS][KH];   // input rows, later reused for f
  __shared__ float ts[ROWS][KH];   // relu(h W1 + b1)
  __shared__ float rn[ROWS];       // 1 / max(||f||, 1e-12)

  const float* h = blockIdx.y ? h1 : h0;
  _Float16*   fh = blockIdx.y ? f1h : f0h;
  const int j    = threadIdx.x;          // 0..319
  const int row0 = blockIdx.x * ROWS;

  for (int r = 0; r < ROWS; ++r)
    hs[r][j] = (j < EMB) ? h[(row0 + r) * EMB + j] : 0.0f;
  __syncthreads();

  // layer 1 + ReLU: W1 column j, broadcast hs[r][k] from LDS
  {
    float acc[ROWS];
    const float bj = (j < EMB) ? b1[j] : 0.0f;
    #pragma unroll
    for (int r = 0; r < ROWS; ++r) acc[r] = bj;
    for (int k = 0; k < EMB; ++k) {
      const float w = (j < EMB) ? W1[k * EMB + j] : 0.0f;
      #pragma unroll
      for (int r = 0; r < ROWS; ++r) acc[r] = fmaf(hs[r][k], w, acc[r]);
    }
    #pragma unroll
    for (int r = 0; r < ROWS; ++r) ts[r][j] = fmaxf(acc[r], 0.0f);
  }
  __syncthreads();

  // layer 2, write f back into hs (hs is dead after layer 1)
  {
    float acc[ROWS];
    const float bj = (j < EMB) ? b2[j] : 0.0f;
    #pragma unroll
    for (int r = 0; r < ROWS; ++r) acc[r] = bj;
    for (int k = 0; k < EMB; ++k) {
      const float w = (j < EMB) ? W2[k * EMB + j] : 0.0f;
      #pragma unroll
      for (int r = 0; r < ROWS; ++r) acc[r] = fmaf(ts[r][k], w, acc[r]);
    }
    #pragma unroll
    for (int r = 0; r < ROWS; ++r) hs[r][j] = (j < EMB) ? acc[r] : 0.0f;
  }
  __syncthreads();

  if (j < ROWS) {
    float s = 0.0f;
    for (int k = 0; k < EMB; ++k) { const float v = hs[j][k]; s = fmaf(v, v, s); }
    rn[j] = 1.0f / fmaxf(sqrtf(s), 1e-12f);
  }
  __syncthreads();

  for (int r = 0; r < ROWS; ++r)
    fh[(row0 + r) * KH + j] = (_Float16)(hs[r][j] * rn[r]);  // j>=300 slots are 0
}

// ---------------- Kernel 2: s0 = exp(A B^T / T) + eps via WMMA -------------
// Each wave: 2x2 tiles of 16x16 (32x32 patch). Block = 8 waves -> 64x128.
__device__ __forceinline__ v16h load_frag(const _Float16* __restrict__ p) {
  v16h f;
  reinterpret_cast<float4*>(&f)[0] = *reinterpret_cast<const float4*>(p);      // K = kb+0..7
  reinterpret_cast<float4*>(&f)[1] = *reinterpret_cast<const float4*>(p + 16); // K = kb+16..23
  return f;
}

__global__ __launch_bounds__(256) void gemm_exp_kernel(
    const _Float16* __restrict__ A,   // f0h [4096][320]
    const _Float16* __restrict__ B,   // f1h [4096][320]
    float* __restrict__ out)          // s0  [4096][4096]
{
  const int wave = threadIdx.x >> 5;
  const int lane = threadIdx.x & 31;
  const int l15  = lane & 15;
  const int kb   = (lane >> 4) * 8;     // 16-bit A/B fragment: kbase 0 or 8

  const int mBase = blockIdx.y * 64 + (wave >> 2) * 32;
  const int nBase = blockIdx.x * 128 + (wave & 3) * 32;

  const _Float16* a0p = A + (mBase      + l15) * KH + kb;
  const _Float16* a1p = A + (mBase + 16 + l15) * KH + kb;
  const _Float16* b0p = B + (nBase      + l15) * KH + kb;
  const _Float16* b1p = B + (nBase + 16 + l15) * KH + kb;

  v8f acc[2][2] = {};
  #pragma unroll
  for (int kk = 0; kk < KH; kk += 32) {
    const v16h a0 = load_frag(a0p + kk);
    const v16h a1 = load_frag(a1p + kk);
    const v16h b0 = load_frag(b0p + kk);
    const v16h b1 = load_frag(b1p + kk);
    acc[0][0] = __builtin_amdgcn_wmma_f32_16x16x32_f16(false, a0, false, b0, (short)0, acc[0][0], false, false);
    acc[0][1] = __builtin_amdgcn_wmma_f32_16x16x32_f16(false, a0, false, b1, (short)0, acc[0][1], false, false);
    acc[1][0] = __builtin_amdgcn_wmma_f32_16x16x32_f16(false, a1, false, b0, (short)0, acc[1][0], false, false);
    acc[1][1] = __builtin_amdgcn_wmma_f32_16x16x32_f16(false, a1, false, b1, (short)0, acc[1][1], false, false);
  }

  // C/D layout: VGPR e -> M = e (lanes 0-15) or e+8 (lanes 16-31); N = lane&15
  #pragma unroll
  for (int mi = 0; mi < 2; ++mi) {
    #pragma unroll
    for (int ni = 0; ni < 2; ++ni) {
      const int rbase = mBase + mi * 16 + ((lane >> 4) << 3);
      const int col   = nBase + ni * 16 + l15;
      #pragma unroll
      for (int e = 0; e < 8; ++e)
        out[(rbase + e) * NN + col] = __expf(acc[mi][ni][e] * 25.0f) + 1e-10f;
    }
  }
}

// ---------------- Sinkhorn helper kernels (bitwise deterministic) ----------
__global__ void init_ones_kernel(float* __restrict__ r) {
  r[blockIdx.x * 256 + threadIdx.x] = 1.0f;
}

// partial[by][j] = sum_{i in 128-row chunk by} s0[i][j] * r[i]
__global__ __launch_bounds__(256) void col_partial_kernel(
    const float* __restrict__ s0, const float* __restrict__ r,
    float* __restrict__ partial)
{
  const int j  = blockIdx.x * 256 + threadIdx.x;
  const int i0 = blockIdx.y * 128;
  float acc = 0.0f;
  for (int i = i0; i < i0 + 128; ++i)
    acc = fmaf(s0[i * NN + j], r[i], acc);
  partial[blockIdx.y * NN + j] = acc;
}

// c[j] = 1 / sum_by partial[by][j]
__global__ __launch_bounds__(256) void col_reduce_kernel(
    const float* __restrict__ partial, float* __restrict__ c)
{
  const int j = blockIdx.x * 256 + threadIdx.x;
  float t = 0.0f;
  #pragma unroll
  for (int by = 0; by < 32; ++by) t += partial[by * NN + j];
  c[j] = 1.0f / t;
}

// r[i] = 1 / sum_j s0[i][j] * c[j]   (one wave per row, fixed reduction tree)
__global__ __launch_bounds__(256) void row_update_kernel(
    const float* __restrict__ s0, const float* __restrict__ c,
    float* __restrict__ r)
{
  const int wave = threadIdx.x >> 5;
  const int lane = threadIdx.x & 31;
  const int i = blockIdx.x * 8 + wave;
  const float4* row = reinterpret_cast<const float4*>(s0 + i * NN);
  const float4* c4  = reinterpret_cast<const float4*>(c);
  float acc = 0.0f;
  for (int j = lane; j < NN / 4; j += 32) {
    const float4 v = row[j];
    const float4 w = c4[j];
    acc += v.x * w.x + v.y * w.y + v.z * w.z + v.w * w.w;
  }
  #pragma unroll
  for (int m = 16; m > 0; m >>= 1) acc += __shfl_xor(acc, m, 32);
  if (lane == 0) r[i] = 1.0f / acc;
}

// out[i][j] = s0[i][j] * r[i] * c[j]  (in place, float4)
__global__ __launch_bounds__(256) void scale_kernel(
    float* __restrict__ out, const float* __restrict__ r,
    const float* __restrict__ c)
{
  const int idx  = blockIdx.x * 256 + threadIdx.x;   // float4 index
  const int base = idx * 4;
  const int i = base >> 12;
  const int j = base & (NN - 1);
  const float ri = r[i];
  float4 v = reinterpret_cast<float4*>(out)[idx];
  v.x *= ri * c[j];
  v.y *= ri * c[j + 1];
  v.z *= ri * c[j + 2];
  v.w *= ri * c[j + 3];
  reinterpret_cast<float4*>(out)[idx] = v;
}

// ---------------------------------------------------------------------------
extern "C" void kernel_launch(void* const* d_in, const int* in_sizes, int n_in,
                              void* d_out, int out_size, void* d_ws, size_t ws_size,
                              hipStream_t stream) {
  (void)in_sizes; (void)n_in; (void)out_size; (void)ws_size;
  const float* h0 = (const float*)d_in[0];
  const float* h1 = (const float*)d_in[1];
  const float* W1 = (const float*)d_in[2];
  const float* b1 = (const float*)d_in[3];
  const float* W2 = (const float*)d_in[4];
  const float* b2 = (const float*)d_in[5];
  float* out = (float*)d_out;

  // workspace carve (~5.8 MB)
  char* ws = (char*)d_ws;
  _Float16* f0h   = (_Float16*)(ws);                           // 4096*320*2 B
  _Float16* f1h   = (_Float16*)(ws + (size_t)NN * KH * 2);     // 4096*320*2 B
  float* rvec     = (float*)(ws + (size_t)NN * KH * 4);        // 4096 f32
  float* cvec     = (float*)(ws + (size_t)NN * KH * 4 + NN * 4);
  float* partial  = (float*)(ws + (size_t)NN * KH * 4 + 2 * NN * 4); // 32*4096 f32

  // 1) projector + l2norm for h0 and h1
  proj_kernel<<<dim3(NN / ROWS, 2), 320, 0, stream>>>(h0, h1, W1, b1, W2, b2, f0h, f1h);

  // 2) s0 = exp(f0 f1^T / 0.04) + 1e-10 straight into d_out (L2-resident)
  gemm_exp_kernel<<<dim3(NN / 128, NN / 64), 256, 0, stream>>>(f0h, f1h, out);

  // 3) Sinkhorn as diagonal scalings
  init_ones_kernel<<<NN / 256, 256, 0, stream>>>(rvec);
  for (int it = 0; it < 10; ++it) {
    if ((it & 1) == 0) {
      col_partial_kernel<<<dim3(NN / 256, 32), 256, 0, stream>>>(out, rvec, partial);
      col_reduce_kernel<<<NN / 256, 256, 0, stream>>>(partial, cvec);
    } else {
      row_update_kernel<<<NN / 8, 256, 0, stream>>>(out, cvec, rvec);
    }
  }

  // 4) out = s0 .* (r x c)
  scale_kernel<<<(NN * (NN / 4)) / 256, 256, 0, stream>>>(out, rvec, cvec);
}
